// Linear_82051055223324
// MI455X (gfx1250) — matrix operand order repaired and checked by reading, not compiled, not run on hardware
//
#include <hip/hip_runtime.h>

// --- WMMA vector types (CDNA5 / gfx1250, wave32) ---
typedef __attribute__((ext_vector_type(16))) __bf16         v16bf;
typedef __attribute__((ext_vector_type(16))) unsigned short v16us;
typedef __attribute__((ext_vector_type(8)))  float          v8f;

// Split fp32 into bf16 hi (top 16 bits, truncated) + bf16 lo (residual).
__device__ __forceinline__ void splitf(float x, unsigned short& h, unsigned short& l) {
    union { float f; unsigned u; } a, hi, r;
    a.f  = x;
    hi.u = a.u & 0xffff0000u;
    r.f  = x - hi.f;
    h = (unsigned short)(hi.u >> 16);
    l = (unsigned short)(r.u  >> 16);
}

// ---------------------------------------------------------------------------
// Pass 1: elementwise fp32 -> (bf16 hi, bf16 lo) planes, optional threshold
// prune fused in. Pure bandwidth: 128MB read + 128MB written ~= 11us @ 23.3TB/s.
// ---------------------------------------------------------------------------
__global__ __launch_bounds__(256) void split_planes(
    const float* __restrict__ src, unsigned short* __restrict__ hi,
    unsigned short* __restrict__ lo, const float* __restrict__ thr_p, int prune) {
    const float thr = prune ? *thr_p : -1.0f;   // |x| > -1 is always true
    const size_t i = ((size_t)blockIdx.x * 256 + threadIdx.x) * 8;
    float4 a = *(const float4*)(src + i);
    float4 b = *(const float4*)(src + i + 4);
    float v[8] = {a.x, a.y, a.z, a.w, b.x, b.y, b.z, b.w};
    union { unsigned short s[8]; uint4 q; } h, l;
#pragma unroll
    for (int e = 0; e < 8; ++e) {
        float xv = v[e];
        xv = (__builtin_fabsf(xv) > thr) ? xv : 0.0f;
        splitf(xv, h.s[e], l.s[e]);
    }
    *(uint4*)(hi + i) = h.q;
    *(uint4*)(lo + i) = l.q;
}

// Load a 16-element bf16 fragment as two 16B runs (ISA 7.12.2 layouts).
__device__ __forceinline__ v16bf load_frag16(const unsigned short* p, int second_off) {
    union { uint4 q[2]; v16bf v; } u;
    u.q[0] = *(const uint4*)(p);
    u.q[1] = *(const uint4*)(p + second_off);
    return u.v;
}

// ---------------------------------------------------------------------------
// Pass 2: GEMM from pre-split bf16 planes.
// Block = 256 threads (8 waves, 2x4). Block tile 128x256; wave tile 64x64
// (4x4 wmma tiles). K-step 32 -> 48 v_wmma_f32_16x16x32_bf16 vs 32 b128
// loads per step (48 FLOP/B from L1/L2). Terms emitted hh, lh, hl so each
// accumulator's dependent WMMAs are separated by 15 independent ones.
// ---------------------------------------------------------------------------
__global__ __launch_bounds__(256) void rtp_gemm_bf16(
    const unsigned short* __restrict__ Xh, const unsigned short* __restrict__ Xl,
    const unsigned short* __restrict__ Wh, const unsigned short* __restrict__ Wl,
    float* __restrict__ Y) {
    constexpr int K = 4096, LD = 4096;

    const int lane    = threadIdx.x & 31;
    const int wave    = threadIdx.x >> 5;
    const int half    = lane >> 4;
    const int ln      = lane & 15;
    const int waveRow = wave & 1;    // 2 row-groups of 64 rows
    const int waveCol = wave >> 1;   // 4 col-groups of 64 cols
    const int rowBase = blockIdx.y * 128 + waveRow * 64;
    const int colBase = blockIdx.x * 256 + waveCol * 64;

    v8f acc[4][4] = {};

    for (int k = 0; k < K; k += 32) {
        // A fragments: lane = row m; elems e<8 -> k+half*8+e, e>=8 -> +16
        v16bf ah[4], al[4];
#pragma unroll
        for (int i = 0; i < 4; ++i) {
            const size_t base = (size_t)(rowBase + i * 16 + ln) * LD + k + half * 8;
            ah[i] = load_frag16(Xh + base, 16);
            al[i] = load_frag16(Xl + base, 16);
        }
        // B fragments: lane = col o; elems e<8 -> k+half*8+e, e>=8 -> +16
        v16bf bh[4], bl[4];
#pragma unroll
        for (int j = 0; j < 4; ++j) {
            const size_t base = (size_t)(colBase + j * 16 + ln) * LD + k + half * 8;
            bh[j] = load_frag16(Wh + base, 16);
            bl[j] = load_frag16(Wl + base, 16);
        }
        // term 1: xh * wh
#pragma unroll
        for (int i = 0; i < 4; ++i)
#pragma unroll
            for (int j = 0; j < 4; ++j)
                acc[i][j] = __builtin_amdgcn_wmma_f32_16x16x32_bf16(
                    false, ah[i], false, bh[j], (short)0, acc[i][j], false, false);
        // term 2: xl * wh
#pragma unroll
        for (int i = 0; i < 4; ++i)
#pragma unroll
            for (int j = 0; j < 4; ++j)
                acc[i][j] = __builtin_amdgcn_wmma_f32_16x16x32_bf16(
                    false, al[i], false, bh[j], (short)0, acc[i][j], false, false);
        // term 3: xh * wl
#pragma unroll
        for (int i = 0; i < 4; ++i)
#pragma unroll
            for (int j = 0; j < 4; ++j)
                acc[i][j] = __builtin_amdgcn_wmma_f32_16x16x32_bf16(
                    false, ah[i], false, bl[j], (short)0, acc[i][j], false, false);
    }

    // C/D layout: VGPR r -> row r + 8*half, col = ln
#pragma unroll
    for (int i = 0; i < 4; ++i) {
#pragma unroll
        for (int j = 0; j < 4; ++j) {
            float* yp = Y + (size_t)(rowBase + i * 16 + half * 8) * LD
                          + colBase + j * 16 + ln;
#pragma unroll
            for (int r = 0; r < 8; ++r)
                yp[(size_t)r * LD] = acc[i][j][r];
        }
    }
}

// ---------------------------------------------------------------------------
// Fallback: fused convert-in-loop GEMM (32x64 wave tile) if ws is too small.
// ---------------------------------------------------------------------------
__global__ __launch_bounds__(256) void rtp_gemm_fused(
    const float* __restrict__ X, const float* __restrict__ W,
    const float* __restrict__ thr_p, float* __restrict__ Y) {
    constexpr int K = 4096, LD = 4096;
    const float thr   = *thr_p;
    const int lane    = threadIdx.x & 31;
    const int wave    = threadIdx.x >> 5;
    const int half    = lane >> 4;
    const int ln      = lane & 15;
    const int waveRow = wave & 3;
    const int waveCol = wave >> 2;
    const int rowBase = blockIdx.y * 128 + waveRow * 32;
    const int colBase = blockIdx.x * 128 + waveCol * 64;

    v8f acc[2][4] = {};

    for (int k = 0; k < K; k += 32) {
        v16bf ah[2], al[2];
#pragma unroll
        for (int i = 0; i < 2; ++i) {
            const float* ap = X + (size_t)(rowBase + i * 16 + ln) * LD + k + half * 8;
            float4 a0 = *(const float4*)(ap);
            float4 a1 = *(const float4*)(ap + 4);
            float4 a2 = *(const float4*)(ap + 16);
            float4 a3 = *(const float4*)(ap + 20);
            float v[16] = {a0.x, a0.y, a0.z, a0.w, a1.x, a1.y, a1.z, a1.w,
                           a2.x, a2.y, a2.z, a2.w, a3.x, a3.y, a3.z, a3.w};
            v16us h, l;
#pragma unroll
            for (int e = 0; e < 16; ++e) {
                float xv = v[e];
                xv = (__builtin_fabsf(xv) > thr) ? xv : 0.0f;
                unsigned short hs, ls; splitf(xv, hs, ls);
                h[e] = hs; l[e] = ls;
            }
            ah[i] = __builtin_bit_cast(v16bf, h);
            al[i] = __builtin_bit_cast(v16bf, l);
        }
        v16bf bh[4], bl[4];
#pragma unroll
        for (int j = 0; j < 4; ++j) {
            const float* bp = W + (size_t)(colBase + j * 16 + ln) * LD + k + half * 8;
            float4 b0 = *(const float4*)(bp);
            float4 b1 = *(const float4*)(bp + 4);
            float4 b2 = *(const float4*)(bp + 16);
            float4 b3 = *(const float4*)(bp + 20);
            float v[16] = {b0.x, b0.y, b0.z, b0.w, b1.x, b1.y, b1.z, b1.w,
                           b2.x, b2.y, b2.z, b2.w, b3.x, b3.y, b3.z, b3.w};
            v16us h, l;
#pragma unroll
            for (int e = 0; e < 16; ++e) {
                unsigned short hs, ls; splitf(v[e], hs, ls);
                h[e] = hs; l[e] = ls;
            }
            bh[j] = __builtin_bit_cast(v16bf, h);
            bl[j] = __builtin_bit_cast(v16bf, l);
        }
#pragma unroll
        for (int i = 0; i < 2; ++i) {
#pragma unroll
            for (int j = 0; j < 4; ++j) {
                acc[i][j] = __builtin_amdgcn_wmma_f32_16x16x32_bf16(
                    false, ah[i], false, bh[j], (short)0, acc[i][j], false, false);
                acc[i][j] = __builtin_amdgcn_wmma_f32_16x16x32_bf16(
                    false, al[i], false, bh[j], (short)0, acc[i][j], false, false);
                acc[i][j] = __builtin_amdgcn_wmma_f32_16x16x32_bf16(
                    false, ah[i], false, bl[j], (short)0, acc[i][j], false, false);
            }
        }
    }

#pragma unroll
    for (int i = 0; i < 2; ++i) {
#pragma unroll
        for (int j = 0; j < 4; ++j) {
            float* yp = Y + (size_t)(rowBase + i * 16 + half * 8) * LD
                          + colBase + j * 16 + ln;
#pragma unroll
            for (int r = 0; r < 8; ++r)
                yp[(size_t)r * LD] = acc[i][j][r];
        }
    }
}

extern "C" void kernel_launch(void* const* d_in, const int* in_sizes, int n_in,
                              void* d_out, int out_size, void* d_ws, size_t ws_size,
                              hipStream_t stream) {
    (void)in_sizes; (void)n_in; (void)out_size;
    const float* X   = (const float*)d_in[0];   // last_out [4096,4096] f32
    const float* W   = (const float*)d_in[1];   // weight   [4096,4096] f32
    const float* thr = (const float*)d_in[2];   // threshold scalar
    float* Y = (float*)d_out;

    const size_t PLANE = (size_t)4096 * 4096;   // elements per plane
    if (ws_size >= 4 * PLANE * sizeof(unsigned short)) {
        unsigned short* Xh = (unsigned short*)d_ws;
        unsigned short* Xl = Xh + PLANE;
        unsigned short* Wh = Xl + PLANE;
        unsigned short* Wl = Wh + PLANE;
        const int nblk = (int)(PLANE / (8 * 256));          // 8 elems/thread
        split_planes<<<nblk, 256, 0, stream>>>(X, Xh, Xl, thr, 1);
        split_planes<<<nblk, 256, 0, stream>>>(W, Wh, Wl, thr, 0);
        const dim3 grid(4096 / 256, 4096 / 128);            // 16 x 32 blocks
        rtp_gemm_bf16<<<grid, dim3(256), 0, stream>>>(Xh, Xl, Wh, Wl, Y);
    } else {
        const dim3 grid(4096 / 128, 4096 / 128);            // 32 x 32 blocks
        rtp_gemm_fused<<<grid, dim3(256), 0, stream>>>(X, W, thr, Y);
    }
}
